// ann2_snn1_6176162971992
// MI455X (gfx1250) — matrix operand order repaired and checked by reading, not compiled, not run on hardware
//
#include <hip/hip_runtime.h>
#include <hip/hip_bf16.h>
#include <math.h>

// ---------------------------------------------------------------------------
// SNN forward on MI455X (gfx1250, wave32, WMMA).
//
// Math reduction: psp[t] = c[t,d] * sigmoid(x[b,d])   (linear IIR, const input)
//   c[t] = a1*c[t-1] + a2*c[t-2] + 1
// => I[t,b,n] = coding[b,:] . (c[t,:]*W[n,:]) + bias[n]
// => one GEMM: coding[2048x512] @ F^T, F[(t*10+n), d] = c[t,d]*W[n,d], [1280x512]
// Then a trivially parallel LIF recurrence over t per (b,n).
// ---------------------------------------------------------------------------

typedef __attribute__((ext_vector_type(16))) _Float16 v16h;
typedef __attribute__((ext_vector_type(8)))  _Float16 v8h;
typedef __attribute__((ext_vector_type(8)))  float    v8f;

#define B_SZ   2048
#define D_RAW  500
#define D_PAD  512
#define NCLS   10
#define T_LEN  128
#define NCOL   (T_LEN * NCLS)   // 1280 GEMM columns (t*10+n)

// workspace layout (bytes), all 256B aligned
#define OFF_CODING 0u
#define SZ_CODING  (B_SZ * D_PAD * 2u)        // 2,097,152  f16 [2048,512]
#define OFF_C      (OFF_CODING + SZ_CODING)
#define SZ_C       (T_LEN * D_PAD * 4u)       //   262,144  f32 [128,512]
#define OFF_F      (OFF_C + SZ_C)
#define SZ_F       (NCOL * D_PAD * 2u)        // 1,310,720  f16 [1280,512]
#define OFF_GEMM   (OFF_F + SZ_F)
#define SZ_GEMM    (B_SZ * NCOL * 4u)         // 10,485,760 f32 [2048,1280]
// total ~14.2 MB

// ---- 1) coding = sigmoid(x), f16, K padded 500->512 with zeros -------------
__global__ void k_sigmoid(const float* __restrict__ x,
                          _Float16* __restrict__ coding) {
  int idx = blockIdx.x * blockDim.x + threadIdx.x;   // 2048*512 threads
  int d = idx & (D_PAD - 1);
  int b = idx >> 9;
  float v = 0.0f;
  if (d < D_RAW) {
    float xv = x[b * D_RAW + d];
    v = 1.0f / (1.0f + expf(-xv));
  }
  coding[idx] = (_Float16)v;
}

// ---- 2) dual-exp IIR impulse-accumulation coefficient c[t,d] ---------------
__global__ void k_iir(const float* __restrict__ a1v,
                      const float* __restrict__ a2v,
                      float* __restrict__ cbuf) {
  int d = threadIdx.x;                                // one block, 512 threads
  float a1 = (d < D_RAW) ? a1v[d] : 0.0f;
  float a2 = (d < D_RAW) ? a2v[d] : 0.0f;
  float p1 = 0.0f, p2 = 0.0f;
  for (int t = 0; t < T_LEN; ++t) {
    float nw = fmaf(a1, p1, fmaf(a2, p2, 1.0f));
    cbuf[t * D_PAD + d] = nw;
    p2 = p1; p1 = nw;
  }
}

// ---- 3) F[(t*10+n), d] = c[t,d] * W[n,d]  (f16, padded) --------------------
__global__ void k_makeF(const float* __restrict__ cbuf,
                        const float* __restrict__ W,
                        _Float16* __restrict__ F) {
  int idx = blockIdx.x * blockDim.x + threadIdx.x;    // 1280*512 threads
  int d   = idx & (D_PAD - 1);
  int row = idx >> 9;
  int t = row / NCLS, n = row % NCLS;
  float w = (d < D_RAW) ? W[n * D_RAW + d] : 0.0f;
  F[idx] = (_Float16)(cbuf[t * D_PAD + d] * w);
}

// ---- 4) WMMA GEMM: I[2048,1280] = coding[2048,512] @ F^T -------------------
// 8 waves/block; wave task = (M 16-tile, group of 4 N 16-tiles). A reused x4.
// A frag (16x32 f16, ISA 7.12.2): lane L -> row M=L%16, halves i:
//   K = kbase + 16*(L>=16) + i(0..15), same order as B   -> one 32B load
// B frag (32x16 f16): lane L -> col N=L%16, K = kbase + 16*(L>=16) + i(0..15)
//   -> one 32B load (F rows are K-contiguous)
// D (16x16 f32, 8 VGPRs): lane L, vgpr r -> (M = r + 8*(L>=16), N = L%16)
__global__ void k_gemm(const _Float16* __restrict__ A,   // coding [2048,512]
                       const _Float16* __restrict__ Bm,  // F      [1280,512]
                       float* __restrict__ Cm) {         // I      [2048,1280]
  const int wave = threadIdx.x >> 5;
  const int lane = threadIdx.x & 31;
  const int task = blockIdx.x * 8 + wave;      // 0..2559 (= 128 mtiles * 20 ngrps)
  const int mtile = task / 20;
  const int ngrp  = task % 20;
  const int m0 = mtile * 16;
  const int n0 = ngrp * 64;

  const int hi = lane >> 4;          // 0: lanes 0-15, 1: lanes 16-31
  const int ln = lane & 15;

  const _Float16* aptr = A + (m0 + ln) * D_PAD + 16 * hi;
  const _Float16* bbase = Bm + 16 * hi;

  v8f acc0 = {}, acc1 = {}, acc2 = {}, acc3 = {};

  for (int k = 0; k < D_PAD; k += 32) {
    union { v16h v; v8h h[2]; } a;
    a.v = *(const v16h*)(aptr + k);

    v16h b0 = *(const v16h*)(bbase + (n0 +  0 + ln) * D_PAD + k);
    v16h b1 = *(const v16h*)(bbase + (n0 + 16 + ln) * D_PAD + k);
    v16h b2 = *(const v16h*)(bbase + (n0 + 32 + ln) * D_PAD + k);
    v16h b3 = *(const v16h*)(bbase + (n0 + 48 + ln) * D_PAD + k);

    acc0 = __builtin_amdgcn_wmma_f32_16x16x32_f16(false, a.v, false, b0,
                                                  (short)0, acc0, false, false);
    acc1 = __builtin_amdgcn_wmma_f32_16x16x32_f16(false, a.v, false, b1,
                                                  (short)0, acc1, false, false);
    acc2 = __builtin_amdgcn_wmma_f32_16x16x32_f16(false, a.v, false, b2,
                                                  (short)0, acc2, false, false);
    acc3 = __builtin_amdgcn_wmma_f32_16x16x32_f16(false, a.v, false, b3,
                                                  (short)0, acc3, false, false);
  }

  float* crow = Cm + (m0 + 8 * hi) * NCOL + n0 + ln;
#pragma unroll
  for (int r = 0; r < 8; ++r) {
    crow[r * NCOL +  0] = acc0[r];
    crow[r * NCOL + 16] = acc1[r];
    crow[r * NCOL + 32] = acc2[r];
    crow[r * NCOL + 48] = acc3[r];
  }
}

// ---- 5) LIF recurrence with reset-to-zero, writes spikes [B, NCLS, T] ------
__global__ void k_lif(const float* __restrict__ I,       // [2048,1280]
                      const float* __restrict__ bias,    // [10]
                      const float* __restrict__ decay,   // [10]
                      float* __restrict__ out) {         // [2048,10,128]
  int tid = blockIdx.x * blockDim.x + threadIdx.x;       // 20480 threads
  int b = tid / NCLS, n = tid % NCLS;
  float bb  = bias[n];
  float dec = decay[n];
  float v = 0.0f, s = 0.0f;
  const float* ip = I + b * NCOL + n;
  float* op = out + (b * NCLS + n) * T_LEN;
  for (int t = 0; t < T_LEN; ++t) {
    float cur = ip[t * NCLS] + bb;
    v = dec * v * (1.0f - s) + cur;
    s = (v >= 1.0f) ? 1.0f : 0.0f;
    op[t] = s;
  }
}

// ---------------------------------------------------------------------------
extern "C" void kernel_launch(void* const* d_in, const int* in_sizes, int n_in,
                              void* d_out, int out_size, void* d_ws, size_t ws_size,
                              hipStream_t stream) {
  const float* x       = (const float*)d_in[0];   // [2048*500]
  const float* alpha_1 = (const float*)d_in[1];   // [500]
  const float* alpha_2 = (const float*)d_in[2];   // [500]
  const float* W       = (const float*)d_in[3];   // [10*500]
  const float* bias    = (const float*)d_in[4];   // [10]
  const float* decay_v = (const float*)d_in[5];   // [10]
  float* out = (float*)d_out;                     // [2048*10*128]

  char* ws = (char*)d_ws;
  _Float16* coding = (_Float16*)(ws + OFF_CODING);
  float*    cbuf   = (float*)   (ws + OFF_C);
  _Float16* F      = (_Float16*)(ws + OFF_F);
  float*    I      = (float*)   (ws + OFF_GEMM);

  // 1) sigmoid -> f16 padded          (2048*512 / 256 = 4096 blocks)
  k_sigmoid<<<4096, 256, 0, stream>>>(x, coding);
  // 2) IIR coefficients c[t,d]        (one block, 512 threads)
  k_iir<<<1, 512, 0, stream>>>(alpha_1, alpha_2, cbuf);
  // 3) F matrix                       (1280*512 / 256 = 2560 blocks)
  k_makeF<<<2560, 256, 0, stream>>>(cbuf, W, F);
  // 4) WMMA GEMM                      (2560 wave-tasks / 8 = 320 blocks)
  k_gemm<<<320, 256, 0, stream>>>(coding, F, I);
  // 5) LIF recurrence + spike output  (20480 / 256 = 80 blocks)
  k_lif<<<80, 256, 0, stream>>>(I, bias, decay_v, out);
}
